// CausalSelfAttention_90134183674068
// MI455X (gfx1250) — matrix operand order repaired and checked by reading, not compiled, not run on hardware
//
#include <hip/hip_runtime.h>
#include <hip/hip_bf16.h>
#include <stdint.h>

// ---------------------------------------------------------------------------
// Causal self-attention for MI455X (gfx1250, wave32, WMMA).
// All GEMMs run as v_wmma_f32_16x16x32_bf16 with f32 accumulation.
// Layouts are chosen so every WMMA A/B fragment is a contiguous 16B load:
//   - weights pre-transposed to [N][K] bf16
//   - Q,K stored [B,H,T,Dh]; V stored transposed [B,H,Dh,T]
// Attention P-tile is re-laid out C->A fragment through LDS.
// ---------------------------------------------------------------------------

typedef __bf16 bf16_t;
typedef __attribute__((ext_vector_type(16))) __bf16 v16bf;
typedef __attribute__((ext_vector_type(8)))  __bf16 v8bf;
typedef __attribute__((ext_vector_type(8)))  float  v8f;

#define D_MODEL 1024
#define N_HEAD  16
#define HEAD_DIM 64
#define SEQ_T   2048
#define BATCH   2
#define M_TOK   (BATCH * SEQ_T)   // 4096 token rows

// ---- helpers ---------------------------------------------------------------

__device__ __forceinline__ bf16_t f2bf(float f) {
  union { float f; uint32_t u; } c; c.f = f;
  uint32_t u = c.u + 0x7FFFu + ((c.u >> 16) & 1u);   // round-to-nearest-even
  union { uint16_t s; bf16_t b; } r; r.s = (uint16_t)(u >> 16);
  return r.b;
}

__device__ __forceinline__ v8f vzero8() {
  v8f z;
#pragma unroll
  for (int i = 0; i < 8; ++i) z[i] = 0.0f;
  return z;
}

__device__ __forceinline__ v8f wmma_bf16(v16bf a, v16bf b, v8f c) {
  // 8 args: (neg_a, A, neg_b, B, c_mod, C, reuse_a, reuse_b)
  return __builtin_amdgcn_wmma_f32_16x16x32_bf16(false, a, false, b,
                                                 (short)0, c, false, false);
}

// A-matrix 16x32 bf16 fragment from a row-major tile (tile origin = base).
// lane l: row = l&15 ; lo 8 elems at k = 8*(l>>4), hi 8 at k = 16 + 8*(l>>4).
__device__ __forceinline__ v16bf load_frag_a(const bf16_t* __restrict__ base,
                                             int ld, int lane) {
  const int r = lane & 15, h = lane >> 4;
  union { v16bf v; v8bf h2[2]; } u;
  u.h2[0] = *(const v8bf*)(base + (size_t)r * ld + 8 * h);
  u.h2[1] = *(const v8bf*)(base + (size_t)r * ld + 16 + 8 * h);
  return u.v;
}

// B-matrix 32x16 bf16 fragment from a transposed (column-major-for-B, i.e.
// Bt[n][k]) row-major tile. lane l: col n = l&15 ; lo 8 elems at k = 8*(l>>4), hi 8 at k = 16 + 8*(l>>4).
__device__ __forceinline__ v16bf load_frag_b(const bf16_t* __restrict__ baseT,
                                             int ld, int lane) {
  const int n = lane & 15, h = lane >> 4;
  union { v16bf v; v8bf h2[2]; } u;
  u.h2[0] = *(const v8bf*)(baseT + (size_t)n * ld + 8 * h);
  u.h2[1] = *(const v8bf*)(baseT + (size_t)n * ld + 16 + 8 * h);
  return u.v;
}

// ---- stage 0: conversions --------------------------------------------------

__global__ void k_cvt_bf16(const float* __restrict__ src,
                           bf16_t* __restrict__ dst, int n) {
  int i = blockIdx.x * blockDim.x + threadIdx.x;
  if (i < n) dst[i] = f2bf(src[i]);
}

// dst[n][k] = src[k][n] ; src is rows x cols (K x N) f32, dst is N x K bf16.
__global__ void k_transpose_bf16(const float* __restrict__ src,
                                 bf16_t* __restrict__ dst,
                                 int rows, int cols) {
  int n = blockIdx.x * blockDim.x + threadIdx.x;
  int k = blockIdx.y;
  if (n < cols) dst[(size_t)n * rows + k] = f2bf(src[(size_t)k * cols + n]);
}

// ---- stage 1: QKV GEMM -----------------------------------------------------
// qkv[m][n] = sum_k xb[m][k] * wT[n][k] + bias[n]
// Scatter into Q,K [B,H,T,Dh] and V-transposed [B,H,Dh,T] (all bf16).

__global__ __launch_bounds__(128)
void k_qkv_gemm(const bf16_t* __restrict__ xb,
                const bf16_t* __restrict__ wT,
                const float*  __restrict__ bias,
                bf16_t* __restrict__ Qb,
                bf16_t* __restrict__ Kb,
                bf16_t* __restrict__ Vt) {
  const int lane = threadIdx.x & 31;
  const int wv   = threadIdx.x >> 5;
  const int n0   = blockIdx.x * 64;
  const int m0   = blockIdx.y * 256 + wv * 64;

  v8f acc[4][4];
#pragma unroll
  for (int i = 0; i < 4; ++i)
#pragma unroll
    for (int j = 0; j < 4; ++j) acc[i][j] = vzero8();

  for (int k0 = 0; k0 < D_MODEL; k0 += 32) {
    v16bf a[4], b[4];
#pragma unroll
    for (int i = 0; i < 4; ++i)
      a[i] = load_frag_a(xb + (size_t)(m0 + 16 * i) * D_MODEL + k0, D_MODEL, lane);
#pragma unroll
    for (int j = 0; j < 4; ++j)
      b[j] = load_frag_b(wT + (size_t)(n0 + 16 * j) * D_MODEL + k0, D_MODEL, lane);
#pragma unroll
    for (int i = 0; i < 4; ++i)
#pragma unroll
      for (int j = 0; j < 4; ++j)
        acc[i][j] = wmma_bf16(a[i], b[j], acc[i][j]);
  }

  const int cn = lane & 15, rh = lane >> 4;
#pragma unroll
  for (int i = 0; i < 4; ++i) {
#pragma unroll
    for (int jj = 0; jj < 4; ++jj) {
      const int nn = n0 + jj * 16 + cn;
      const float bv = bias[nn];
#pragma unroll
      for (int j = 0; j < 8; ++j) {
        const int mm = m0 + i * 16 + rh * 8 + j;
        const int bb = mm >> 11, t = mm & (SEQ_T - 1);
        const bf16_t v = f2bf(acc[i][jj][j] + bv);
        if (nn < D_MODEL) {
          const int h = nn >> 6, d = nn & 63;
          Qb[(((size_t)bb * N_HEAD + h) * SEQ_T + t) * HEAD_DIM + d] = v;
        } else if (nn < 2 * D_MODEL) {
          const int c = nn - D_MODEL, h = c >> 6, d = c & 63;
          Kb[(((size_t)bb * N_HEAD + h) * SEQ_T + t) * HEAD_DIM + d] = v;
        } else {
          const int c = nn - 2 * D_MODEL, h = c >> 6, d = c & 63;
          Vt[(((size_t)bb * N_HEAD + h) * HEAD_DIM + d) * SEQ_T + t] = v;
        }
      }
    }
  }
}

// ---- stage 2: flash attention ---------------------------------------------
// One wave = 16 query rows. Streams 32-key tiles with online softmax.

__global__ __launch_bounds__(128)
void k_attn(const bf16_t* __restrict__ Qb,
            const bf16_t* __restrict__ Kb,
            const bf16_t* __restrict__ Vt,
            bf16_t* __restrict__ attout) {
  __shared__ __align__(16) bf16_t pSh[4][16][40];   // per-wave P tile, padded

  const int lane  = threadIdx.x & 31;
  const int wv    = threadIdx.x >> 5;
  const int bh    = blockIdx.y;                 // 0..31  (b*16 + h)
  const int qBase = blockIdx.x * 64 + wv * 16;
  const int cn = lane & 15, rh = lane >> 4;

  const bf16_t* Qp = Qb + ((size_t)bh * SEQ_T + qBase) * HEAD_DIM;
  const bf16_t* Kp = Kb + (size_t)bh * SEQ_T * HEAD_DIM;
  const bf16_t* Vp = Vt + (size_t)bh * HEAD_DIM * SEQ_T;

  const v16bf qa0 = load_frag_a(Qp, HEAD_DIM, lane);        // d = 0..31
  const v16bf qa1 = load_frag_a(Qp + 32, HEAD_DIM, lane);   // d = 32..63

  v8f o[4];
#pragma unroll
  for (int t = 0; t < 4; ++t) o[t] = vzero8();
  float m_r[8], l_r[8];
#pragma unroll
  for (int j = 0; j < 8; ++j) { m_r[j] = -1e30f; l_r[j] = 0.0f; }

  const int qEnd = qBase + 15;
  for (int kb = 0; kb <= qEnd; kb += 32) {
    const bf16_t* K0 = Kp + (size_t)kb * HEAD_DIM;
    // S = Q * K^T   (keys kb..kb+15 -> s0 ; kb+16..kb+31 -> s1)
    v8f s0 = vzero8(), s1 = vzero8();
    s0 = wmma_bf16(qa0, load_frag_b(K0, HEAD_DIM, lane), s0);
    s0 = wmma_bf16(qa1, load_frag_b(K0 + 32, HEAD_DIM, lane), s0);
    s1 = wmma_bf16(qa0, load_frag_b(K0 + 16 * HEAD_DIM, HEAD_DIM, lane), s1);
    s1 = wmma_bf16(qa1, load_frag_b(K0 + 16 * HEAD_DIM + 32, HEAD_DIM, lane), s1);

    const bool needMask = (kb + 31 > qBase);

#pragma unroll
    for (int j = 0; j < 8; ++j) {
      const int qi = qBase + rh * 8 + j;
      float x0 = s0[j] * 0.125f;                 // 1/sqrt(64)
      float x1 = s1[j] * 0.125f;
      if (needMask) {
        if (kb + cn      > qi) x0 = -1e30f;
        if (kb + 16 + cn > qi) x1 = -1e30f;
      }
      float mx = fmaxf(x0, x1);
#pragma unroll
      for (int off = 1; off < 16; off <<= 1)
        mx = fmaxf(mx, __shfl_xor(mx, off, 32));
      const float mnew = fmaxf(m_r[j], mx);
      const float corr = __expf(m_r[j] - mnew);
      const float p0 = __expf(x0 - mnew);
      const float p1 = __expf(x1 - mnew);
      float rs = p0 + p1;
#pragma unroll
      for (int off = 1; off < 16; off <<= 1)
        rs += __shfl_xor(rs, off, 32);
      l_r[j] = l_r[j] * corr + rs;
      m_r[j] = mnew;
#pragma unroll
      for (int t = 0; t < 4; ++t) o[t][j] *= corr;

      const int row = rh * 8 + j;                // C-layout row within tile
      pSh[wv][row][cn]      = f2bf(p0);
      pSh[wv][row][16 + cn] = f2bf(p1);
    }
    asm volatile("s_wait_dscnt 0" ::: "memory");

    // Re-read P as a 16x32 A-fragment (row = lane&15).
    union { v16bf v; v8bf h2[2]; } pa;
    pa.h2[0] = *(const v8bf*)&pSh[wv][cn][8 * rh];
    pa.h2[1] = *(const v8bf*)&pSh[wv][cn][16 + 8 * rh];

    // O += P * V   (V transposed [Dh][T] -> contiguous B fragments)
#pragma unroll
    for (int t = 0; t < 4; ++t)
      o[t] = wmma_bf16(pa.v, load_frag_b(Vp + (size_t)(t * 16) * SEQ_T + kb,
                                         SEQ_T, lane), o[t]);
  }

  // epilogue: normalize and scatter to attout [B*T][C] bf16
  const int b = bh >> 4, h = bh & 15;
#pragma unroll
  for (int j = 0; j < 8; ++j) {
    const float inv = 1.0f / l_r[j];
    const int tok = qBase + rh * 8 + j;
    const size_t rowoff = ((size_t)b * SEQ_T + tok) * D_MODEL + h * HEAD_DIM;
#pragma unroll
    for (int t = 0; t < 4; ++t)
      attout[rowoff + t * 16 + cn] = f2bf(o[t][j] * inv);
  }
}

// ---- stage 3: output projection -------------------------------------------

__global__ __launch_bounds__(128)
void k_proj_gemm(const bf16_t* __restrict__ A,
                 const bf16_t* __restrict__ wT,
                 const float*  __restrict__ bias,
                 float* __restrict__ out) {
  const int lane = threadIdx.x & 31;
  const int wv   = threadIdx.x >> 5;
  const int n0   = blockIdx.x * 64;
  const int m0   = blockIdx.y * 256 + wv * 64;

  v8f acc[4][4];
#pragma unroll
  for (int i = 0; i < 4; ++i)
#pragma unroll
    for (int j = 0; j < 4; ++j) acc[i][j] = vzero8();

  for (int k0 = 0; k0 < D_MODEL; k0 += 32) {
    v16bf a[4], b[4];
#pragma unroll
    for (int i = 0; i < 4; ++i)
      a[i] = load_frag_a(A + (size_t)(m0 + 16 * i) * D_MODEL + k0, D_MODEL, lane);
#pragma unroll
    for (int j = 0; j < 4; ++j)
      b[j] = load_frag_b(wT + (size_t)(n0 + 16 * j) * D_MODEL + k0, D_MODEL, lane);
#pragma unroll
    for (int i = 0; i < 4; ++i)
#pragma unroll
      for (int j = 0; j < 4; ++j)
        acc[i][j] = wmma_bf16(a[i], b[j], acc[i][j]);
  }

  const int cn = lane & 15, rh = lane >> 4;
#pragma unroll
  for (int i = 0; i < 4; ++i)
#pragma unroll
    for (int jj = 0; jj < 4; ++jj) {
      const int nn = n0 + jj * 16 + cn;
      const float bv = bias[nn];
#pragma unroll
      for (int j = 0; j < 8; ++j) {
        const int mm = m0 + i * 16 + rh * 8 + j;
        out[(size_t)mm * D_MODEL + nn] = acc[i][jj][j] + bv;
      }
    }
}

// ---- host launcher ---------------------------------------------------------

extern "C" void kernel_launch(void* const* d_in, const int* in_sizes, int n_in,
                              void* d_out, int out_size, void* d_ws, size_t ws_size,
                              hipStream_t stream) {
  const float* x      = (const float*)d_in[0];
  const float* w_qkv  = (const float*)d_in[1];
  const float* b_qkv  = (const float*)d_in[2];
  const float* w_proj = (const float*)d_in[3];
  const float* b_proj = (const float*)d_in[4];
  float* out = (float*)d_out;

  // workspace carve-out (bf16), total 48 MB
  bf16_t* xb     = (bf16_t*)d_ws;
  bf16_t* wqkvT  = xb     + (size_t)M_TOK * D_MODEL;          //  8 MB
  bf16_t* wprojT = wqkvT  + (size_t)3 * D_MODEL * D_MODEL;    //  6 MB
  bf16_t* Qb     = wprojT + (size_t)D_MODEL * D_MODEL;        //  2 MB
  bf16_t* Kb     = Qb     + (size_t)M_TOK * D_MODEL;          //  8 MB
  bf16_t* Vt     = Kb     + (size_t)M_TOK * D_MODEL;          //  8 MB
  bf16_t* attout = Vt     + (size_t)M_TOK * D_MODEL;          //  8 MB (+8)

  k_cvt_bf16<<<dim3((M_TOK * D_MODEL) / 256), dim3(256), 0, stream>>>(
      x, xb, M_TOK * D_MODEL);
  k_transpose_bf16<<<dim3((3 * D_MODEL) / 256, D_MODEL), dim3(256), 0, stream>>>(
      w_qkv, wqkvT, D_MODEL, 3 * D_MODEL);
  k_transpose_bf16<<<dim3(D_MODEL / 256, D_MODEL), dim3(256), 0, stream>>>(
      w_proj, wprojT, D_MODEL, D_MODEL);

  k_qkv_gemm<<<dim3((3 * D_MODEL) / 64, M_TOK / 256), dim3(128), 0, stream>>>(
      xb, wqkvT, b_qkv, Qb, Kb, Vt);

  k_attn<<<dim3(SEQ_T / 64, BATCH * N_HEAD), dim3(128), 0, stream>>>(
      Qb, Kb, Vt, attout);

  k_proj_gemm<<<dim3(D_MODEL / 64, M_TOK / 256), dim3(128), 0, stream>>>(
      attout, wprojT, b_proj, out);
}
